// HardwareOptimizedAttention_13932873908394
// MI455X (gfx1250) — matrix operand order repaired and checked by reading, not compiled, not run on hardware
//
#include <hip/hip_runtime.h>

// ---------------------------------------------------------------------------
// MI455X (gfx1250) no-softmax attention, bf16 WMMA pipeline, round 4:
//   - GEMMs: 64x64 register tiles (codegen verified good in round 2)
//   - attention: 8 waves/block share one (b,h); K/Vt chunks staged through
//     double-buffered LDS via global_load_async_to_lds_b128 (signature fixed:
//     the builtin takes int4-vector pointers in AS(1)/AS(3))
//   B=2, S=2048, D=2048, H=16, HD=128, BS=B*S=4096
// ---------------------------------------------------------------------------

typedef __attribute__((ext_vector_type(16))) __bf16 v16bf;
typedef __attribute__((ext_vector_type(8)))  __bf16 v8bf;
typedef __attribute__((ext_vector_type(4)))  __bf16 v4bf;
typedef __attribute__((ext_vector_type(8)))  float  v8f;

constexpr int Bn  = 2;
constexpr int Sn  = 2048;
constexpr int Dn  = 2048;
constexpr int Hn  = 16;
constexpr int HDn = 128;
constexpr int BSn = Bn * Sn;   // 4096 tokens

#define DEV_INLINE __device__ __forceinline__

// ---- CDNA5 async global->LDS staging (guarded; fallback = load + ds_store) --
#if defined(__has_builtin)
#if __has_builtin(__builtin_amdgcn_global_load_async_to_lds_b128)
#define HAVE_ASYNC_LDS 1
#endif
#endif
#ifndef HAVE_ASYNC_LDS
#define HAVE_ASYNC_LDS 0
#endif

// b128 payload type as expected by the builtin (per hipcc diagnostic):
//   param0: int4 __device__* (AS1 global), param1: LDS-side pointer (AS3)
typedef int v4i_ls __attribute__((vector_size(4 * sizeof(int))));
typedef __attribute__((address_space(1))) v4i_ls glb_b128_t;
typedef __attribute__((address_space(3))) v4i_ls lds_b128_t;

DEV_INLINE void wait_async_lds() {
#if defined(__has_builtin)
#if __has_builtin(__builtin_amdgcn_s_wait_asynccnt)
  __builtin_amdgcn_s_wait_asynccnt(0);
#else
  asm volatile("s_wait_asynccnt 0x0" ::: "memory");
#endif
#else
  asm volatile("s_wait_asynccnt 0x0" ::: "memory");
#endif
}

DEV_INLINE v8f wmma_bf16(v16bf a, v16bf b, v8f c) {
  // D = A(16x32 bf16) x B(32x16 bf16) + C(16x16 f32)
  return __builtin_amdgcn_wmma_f32_16x16x32_bf16(
      /*neg_a=*/false, a, /*neg_b=*/false, b,
      /*c_mod=*/(short)0, c, /*reuse_a=*/false, /*reuse_b=*/false);
}

// A-matrix fragment (16x32, bf16), source rows K-contiguous with row stride ld.
// Contiguous-k layout used here: lanes 0-15: M=lane, halves[0..15]=K 0..15
//             lanes 16-31: M=lane-16, halves[0..15]=K 16..31
DEV_INLINE v16bf load_afrag(const __bf16* __restrict__ base, int row0, int ld,
                            int k0, int lane) {
  const int r = lane & 15, h = lane >> 4;
  const __bf16* p = base + (size_t)(row0 + r) * ld + (k0 + h * 16);
  v8bf lo = *(const v8bf*)p;
  v8bf hi = *(const v8bf*)(p + 8);
  v16bf o;
#pragma unroll
  for (int i = 0; i < 8; ++i) { o[i] = lo[i]; o[8 + i] = hi[i]; }
  return o;
}

// B-matrix fragment (32x16, bf16); column n is a K-contiguous source row.
// Contiguous-k layout used here: lanes 0-15: N=lane, halves[0..15]=K 0..15
//             lanes 16-31: N=lane-16, halves[0..15]=K 16..31
DEV_INLINE v16bf load_bfrag(const __bf16* __restrict__ base, int n0, int ld,
                            int k0, int lane) {
  const int c = lane & 15, h = lane >> 4;
  return *(const v16bf*)(base + (size_t)(n0 + c) * ld + (k0 + h * 16));
}

// B fragment from LDS in 16B accesses (LDS rows padded to 80B); half h holds K 8h..8h+7, 16+8h..16+8h+7.
DEV_INLINE v16bf lds_bfrag(const __bf16* base, int n0, int ld, int lane) {
  const int c = lane & 15, h = lane >> 4;
  const __bf16* p = base + (n0 + c) * ld + h * 8;
  v8bf l0 = *(const v8bf*)p;
  v8bf l1 = *(const v8bf*)(p + 16);
  v16bf o;
#pragma unroll
  for (int i = 0; i < 8; ++i) { o[i] = l0[i]; o[8 + i] = l1[i]; }
  return o;
}

// ---------------------------------------------------------------------------
// f32 -> bf16 convert (vectorized x4)
// ---------------------------------------------------------------------------
__global__ void __launch_bounds__(256) k_cvt_bf16(const float* __restrict__ in,
                                                  __bf16* __restrict__ out,
                                                  int n4) {
  int i = blockIdx.x * blockDim.x + threadIdx.x;
  if (i < n4) {
    float4 f = ((const float4*)in)[i];
    v4bf o;
    o[0] = (__bf16)f.x; o[1] = (__bf16)f.y; o[2] = (__bf16)f.z; o[3] = (__bf16)f.w;
    ((v4bf*)out)[i] = o;
  }
}

// ---------------------------------------------------------------------------
// Out[token, feat] (bf16) = X[token,:] . W[feat,:]   (y = x @ W^T)
// Swapped convention: A-matrix = W rows (M=feature), B-matrix = X rows (N=token)
// Wave tile 64x64 (16 accs), block = 8 waves -> 128 feat x 256 tokens.
// ---------------------------------------------------------------------------
__global__ void __launch_bounds__(256) k_gemm_wx_bf16(
    const __bf16* __restrict__ W, const __bf16* __restrict__ X,
    __bf16* __restrict__ Out) {
  const int lane = threadIdx.x & 31;
  const int wave = threadIdx.x >> 5;
  const int f0 = blockIdx.y * 128 + (wave & 1) * 64;
  const int t0 = blockIdx.x * 256 + (wave >> 1) * 64;
  v8f acc[4][4] = {};
  for (int k0 = 0; k0 < Dn; k0 += 32) {
    v16bf a[4], bb[4];
#pragma unroll
    for (int i = 0; i < 4; ++i) a[i]  = load_afrag(W, f0 + 16 * i, Dn, k0, lane);
#pragma unroll
    for (int i = 0; i < 4; ++i) bb[i] = load_bfrag(X, t0 + 16 * i, Dn, k0, lane);
#pragma unroll
    for (int fi = 0; fi < 4; ++fi)
#pragma unroll
      for (int ti = 0; ti < 4; ++ti)
        acc[fi][ti] = wmma_bf16(a[fi], bb[ti], acc[fi][ti]);
  }
#pragma unroll
  for (int fi = 0; fi < 4; ++fi)
#pragma unroll
    for (int ti = 0; ti < 4; ++ti) {
      const int tok = t0 + ti * 16 + (lane & 15);
      const int fb  = f0 + fi * 16 + (lane >> 4) * 8;
      v8bf st;
#pragma unroll
      for (int i = 0; i < 8; ++i) st[i] = (__bf16)acc[fi][ti][i];
      *(v8bf*)(Out + (size_t)tok * Dn + fb) = st;
    }
}

// Same GEMM but fp32 output (final projection writes d_out directly).
__global__ void __launch_bounds__(256) k_gemm_wx_f32(
    const __bf16* __restrict__ W, const __bf16* __restrict__ X,
    float* __restrict__ Out) {
  const int lane = threadIdx.x & 31;
  const int wave = threadIdx.x >> 5;
  const int f0 = blockIdx.y * 128 + (wave & 1) * 64;
  const int t0 = blockIdx.x * 256 + (wave >> 1) * 64;
  v8f acc[4][4] = {};
  for (int k0 = 0; k0 < Dn; k0 += 32) {
    v16bf a[4], bb[4];
#pragma unroll
    for (int i = 0; i < 4; ++i) a[i]  = load_afrag(W, f0 + 16 * i, Dn, k0, lane);
#pragma unroll
    for (int i = 0; i < 4; ++i) bb[i] = load_bfrag(X, t0 + 16 * i, Dn, k0, lane);
#pragma unroll
    for (int fi = 0; fi < 4; ++fi)
#pragma unroll
      for (int ti = 0; ti < 4; ++ti)
        acc[fi][ti] = wmma_bf16(a[fi], bb[ti], acc[fi][ti]);
  }
#pragma unroll
  for (int fi = 0; fi < 4; ++fi)
#pragma unroll
    for (int ti = 0; ti < 4; ++ti) {
      const int tok = t0 + ti * 16 + (lane & 15);
      const int fb  = f0 + fi * 16 + (lane >> 4) * 8;
      *(v8f*)(Out + (size_t)tok * Dn + fb) = acc[fi][ti];
    }
}

// ---------------------------------------------------------------------------
// Vt[feat, token] (bf16) = (X @ Wv^T)^T
// ---------------------------------------------------------------------------
__global__ void __launch_bounds__(256) k_gemm_xw_vt(
    const __bf16* __restrict__ X, const __bf16* __restrict__ W,
    __bf16* __restrict__ Vt) {
  const int lane = threadIdx.x & 31;
  const int wave = threadIdx.x >> 5;
  const int f0 = blockIdx.y * 128 + (wave & 1) * 64;   // N side (features)
  const int t0 = blockIdx.x * 256 + (wave >> 1) * 64;  // M side (tokens)
  v8f acc[4][4] = {};                                  // [tokenTile][featTile]
  for (int k0 = 0; k0 < Dn; k0 += 32) {
    v16bf a[4], bb[4];
#pragma unroll
    for (int i = 0; i < 4; ++i) a[i]  = load_afrag(X, t0 + 16 * i, Dn, k0, lane);
#pragma unroll
    for (int i = 0; i < 4; ++i) bb[i] = load_bfrag(W, f0 + 16 * i, Dn, k0, lane);
#pragma unroll
    for (int ti = 0; ti < 4; ++ti)
#pragma unroll
      for (int fi = 0; fi < 4; ++fi)
        acc[ti][fi] = wmma_bf16(a[ti], bb[fi], acc[ti][fi]);
  }
#pragma unroll
  for (int ti = 0; ti < 4; ++ti)
#pragma unroll
    for (int fi = 0; fi < 4; ++fi) {
      const int feat = f0 + fi * 16 + (lane & 15);
      const int tb   = t0 + ti * 16 + (lane >> 4) * 8;
      v8bf st;
#pragma unroll
      for (int i = 0; i < 8; ++i) st[i] = (__bf16)acc[ti][fi][i];
      *(v8bf*)(Vt + (size_t)feat * BSn + tb) = st;
    }
}

// ---------------------------------------------------------------------------
// Attention (no softmax). Block = 8 waves, all on the SAME (b,h): consecutive
// 32-query tiles. Per 32-key chunk, the block cooperatively stages
//   K chunk  (32 keys x 128 hd)  -> lk  (rows padded to 136 halves)
//   Vt chunk (128 hd x 32 keys)  -> lv  (rows padded to 40 halves)
// into double-buffered LDS (async-to-LDS when available), then every wave
// builds its fragments from LDS. scoresT = K.Q^T; its f32 C-layout is
// element-identical to the bf16 16x32 A-fragment needed for scores @ V.
// ---------------------------------------------------------------------------
__global__ void __launch_bounds__(256) k_attn(
    const __bf16* __restrict__ Q, const __bf16* __restrict__ K,
    const __bf16* __restrict__ Vt, const float* __restrict__ mask,
    __bf16* __restrict__ O) {
  __shared__ __bf16 lk[2][32][136];    // 17,408 B  (row shift = 4 banks)
  __shared__ __bf16 lv[2][128][40];    // 20,480 B  (row shift = 20 banks)

  const int tid  = threadIdx.x;
  const int lane = tid & 31;
  const int wave = tid >> 5;
  const int task = blockIdx.x * 8 + wave;   // 8 q-tiles of one (b,h) per block
  const int qt = task & (Sn / 32 - 1);
  const int bh = task >> 6;
  const int h  = bh & (Hn - 1);
  const int b  = bh >> 4;

  const int row0 = b * Sn;
  const __bf16* Qp = Q  + (size_t)row0 * Dn + h * HDn;
  const __bf16* Kp = K  + (size_t)row0 * Dn + h * HDn;
  const __bf16* Vp = Vt + (size_t)(h * HDn) * BSn + row0;     // Vt[hd, key]
  const float*  Mp = mask + (size_t)b * Sn * Sn;
  const int q0 = qt * 32;
  const float scl = 0.08838834764831845f;                     // 1/sqrt(128)

  // Cooperative staging of one 32-key chunk into LDS buffer `sel`.
  auto stage = [&](int kc, int sel) {
    const int kr = tid >> 3, kcol = (tid & 7) * 16;   // K: 32 rows x 128 cols
    const int vr = tid >> 1, vcol = (tid & 1) * 16;   // Vt: 128 rows x 32 cols
    const __bf16* gk = Kp + (size_t)(kc + kr) * Dn + kcol;
    const __bf16* gv = Vp + (size_t)vr * BSn + (kc + vcol);
    __bf16* sk = &lk[sel][kr][kcol];
    __bf16* sv = &lv[sel][vr][vcol];
#if HAVE_ASYNC_LDS
    __builtin_amdgcn_global_load_async_to_lds_b128(
        (glb_b128_t*)gk, (lds_b128_t*)sk, 0, 0);
    __builtin_amdgcn_global_load_async_to_lds_b128(
        (glb_b128_t*)gk, (lds_b128_t*)sk, 16, 0);
    __builtin_amdgcn_global_load_async_to_lds_b128(
        (glb_b128_t*)gv, (lds_b128_t*)sv, 0, 0);
    __builtin_amdgcn_global_load_async_to_lds_b128(
        (glb_b128_t*)gv, (lds_b128_t*)sv, 16, 0);
#else
    *(v8bf*)sk       = *(const v8bf*)gk;
    *(v8bf*)(sk + 8) = *(const v8bf*)(gk + 8);
    *(v8bf*)sv       = *(const v8bf*)gv;
    *(v8bf*)(sv + 8) = *(const v8bf*)(gv + 8);
#endif
  };

  // Q B-fragments register-resident for the whole key loop (32 q x 128 hd).
  v16bf qf[2][4];
#pragma unroll
  for (int u = 0; u < 2; ++u)
#pragma unroll
    for (int c = 0; c < 4; ++c)
      qf[u][c] = load_bfrag(Qp, q0 + u * 16, Dn, c * 32, lane);

  stage(0, 0);

  v8f o[2][8] = {};                                           // 32 q x 128 hd
  for (int kc = 0; kc < Sn; kc += 32) {
    const int sel = (kc >> 5) & 1;
#if HAVE_ASYNC_LDS
    wait_async_lds();          // our staging of buffer `sel` has landed
#endif
    __syncthreads();           // everyone's staging visible; prior reads done
    if (kc + 32 < Sn) stage(kc + 32, sel ^ 1);

    // scoresT: st[t][u] = keys (kc+16t..+15) x queries (q0+16u..+15)
    v8f st[2][2] = {};
#pragma unroll
    for (int c = 0; c < 4; ++c) {
      v16bf ka0 = load_afrag(&lk[sel][0][0], 0,  136, c * 32, lane);
      v16bf ka1 = load_afrag(&lk[sel][0][0], 16, 136, c * 32, lane);
      st[0][0] = wmma_bf16(ka0, qf[0][c], st[0][0]);
      st[0][1] = wmma_bf16(ka0, qf[1][c], st[0][1]);
      st[1][0] = wmma_bf16(ka1, qf[0][c], st[1][0]);
      st[1][1] = wmma_bf16(ka1, qf[1][c], st[1][1]);
    }
    // scale + additive mask, convert to 16x32 bf16 A-fragments (per q sub-tile)
    v16bf sa[2];
#pragma unroll
    for (int u = 0; u < 2; ++u) {
      const int qrow  = q0 + u * 16 + (lane & 15);
      const int kbase = kc + (lane >> 4) * 8;
      const float* m0 = Mp + (size_t)qrow * Sn + kbase;
#pragma unroll
      for (int i = 0; i < 8; ++i) {
        sa[u][i]     = (__bf16)(st[0][u][i] * scl + m0[i]);
        sa[u][8 + i] = (__bf16)(st[1][u][i] * scl + m0[16 + i]);
      }
    }
    // out += scores(16x32) x V(32x128); each V fragment feeds both q sub-tiles
#pragma unroll
    for (int j = 0; j < 8; ++j) {
      v16bf vf = lds_bfrag(&lv[sel][0][0], j * 16, 40, lane);
      o[0][j] = wmma_bf16(sa[0], vf, o[0][j]);
      o[1][j] = wmma_bf16(sa[1], vf, o[1][j]);
    }
  }
  // Store attn_out[token, D] bf16. C-layout: lane -> hd column, 8 q rows.
#pragma unroll
  for (int u = 0; u < 2; ++u)
#pragma unroll
    for (int j = 0; j < 8; ++j) {
      const int hd = j * 16 + (lane & 15);
      const int qb = q0 + u * 16 + (lane >> 4) * 8;
      __bf16* p = O + (size_t)(row0 + qb) * Dn + h * HDn + hd;
#pragma unroll
      for (int i = 0; i < 8; ++i) p[(size_t)i * Dn] = (__bf16)o[u][j][i];
    }
}

// ---------------------------------------------------------------------------
// Launch
// ---------------------------------------------------------------------------
extern "C" void kernel_launch(void* const* d_in, const int* in_sizes, int n_in,
                              void* d_out, int out_size, void* d_ws, size_t ws_size,
                              hipStream_t stream) {
  const float* x    = (const float*)d_in[0];  // [B,S,D]
  const float* mask = (const float*)d_in[1];  // [B,1,S,S]
  const float* wq   = (const float*)d_in[2];  // [D,D]
  const float* wk   = (const float*)d_in[3];
  const float* wv   = (const float*)d_in[4];
  const float* wo   = (const float*)d_in[5];
  float* out = (float*)d_out;                 // [B,S,D] fp32

  // Workspace layout (all bf16): X, 4 weights, Q, K, Vt, attn_out  (~117 MB)
  char* ws = (char*)d_ws;
  const size_t SZ_X = (size_t)BSn * Dn * 2;   // 16 MiB
  const size_t SZ_W = (size_t)Dn * Dn * 2;    //  8 MiB
  __bf16* Xb  = (__bf16*)(ws);
  __bf16* Wqb = (__bf16*)(ws + SZ_X);
  __bf16* Wkb = (__bf16*)(ws + SZ_X + 1 * SZ_W);
  __bf16* Wvb = (__bf16*)(ws + SZ_X + 2 * SZ_W);
  __bf16* Wob = (__bf16*)(ws + SZ_X + 3 * SZ_W);
  __bf16* Qb  = (__bf16*)(ws + 1 * SZ_X + 4 * SZ_W);
  __bf16* Kb  = (__bf16*)(ws + 2 * SZ_X + 4 * SZ_W);
  __bf16* Vtb = (__bf16*)(ws + 3 * SZ_X + 4 * SZ_W);
  __bf16* Ab  = (__bf16*)(ws + 4 * SZ_X + 4 * SZ_W);

  // fp32 -> bf16 conversions
  const int nx4 = BSn * Dn / 4;   // 2,097,152
  const int nw4 = Dn * Dn / 4;    // 1,048,576
  k_cvt_bf16<<<nx4 / 256, 256, 0, stream>>>(x,  Xb,  nx4);
  k_cvt_bf16<<<nw4 / 256, 256, 0, stream>>>(wq, Wqb, nw4);
  k_cvt_bf16<<<nw4 / 256, 256, 0, stream>>>(wk, Wkb, nw4);
  k_cvt_bf16<<<nw4 / 256, 256, 0, stream>>>(wv, Wvb, nw4);
  k_cvt_bf16<<<nw4 / 256, 256, 0, stream>>>(wo, Wob, nw4);

  // Projections: block tile 128 feat x 256 tokens, 8 waves of 64x64
  dim3 gg(BSn / 256, Dn / 128);   // (16, 16)
  k_gemm_wx_bf16<<<gg, 256, 0, stream>>>(Wqb, Xb, Qb);
  k_gemm_wx_bf16<<<gg, 256, 0, stream>>>(Wkb, Xb, Kb);
  k_gemm_xw_vt <<<gg, 256, 0, stream>>>(Xb, Wvb, Vtb);

  // Attention: B*H*(S/32) = 2048 wave-tasks, 8 waves per block
  k_attn<<<(Bn * Hn * (Sn / 32)) / 8, 256, 0, stream>>>(Qb, Kb, Vtb, mask, Ab);

  // Output projection (fp32 result)
  k_gemm_wx_f32<<<gg, 256, 0, stream>>>(Wob, Ab, out);
}
